// GNN_53463752901227
// MI455X (gfx1250) — matrix operand order repaired and checked by reading, not compiled, not run on hardware
//
#include <hip/hip_runtime.h>

typedef __attribute__((ext_vector_type(16))) _Float16 v16h;
typedef __attribute__((ext_vector_type(8)))  float    v8f;

#define BATCH   8
#define NPRED   16320
#define NC      80
#define NCAND   (NPRED * NC)
#define KPRE    300
#define MAXDET  100
#define CAP     1024
#define CONF    0.1f
#define IOUTH   0.6f

// ---------------------------------------------------------------------------
// Kernel 1: per-image two-level float-bit histogram -> score cutoff bits
// ---------------------------------------------------------------------------
__global__ void topk_thresh(const float* __restrict__ pred,
                            unsigned* __restrict__ thresh,
                            unsigned* __restrict__ ccount) {
  const int b = blockIdx.x, tid = threadIdx.x, nt = blockDim.x;
  __shared__ unsigned hist[1024];
  __shared__ int s_coarse;
  __shared__ unsigned s_above, s_thresh;
  const float* po = pred + (size_t)b * NPRED * 85;

  for (int i = tid; i < 1024; i += nt) hist[i] = 0u;
  __syncthreads();
  // pass 1: coarse histogram (bits >> 20) of scores > CONF
  for (int k = tid; k < NCAND; k += nt) {
    int n = k / NC, c = k - n * NC;
    float s = po[n * 85 + 4] * po[n * 85 + 5 + c];
    if (s > CONF) atomicAdd(&hist[__float_as_uint(s) >> 20], 1u);
  }
  __syncthreads();
  if (tid == 0) {
    unsigned cum = 0, above = 0; int cb = -1;
    for (int i = 1023; i >= 0; --i) {
      unsigned h = hist[i];
      if (cum + h >= (unsigned)KPRE) { cb = i; above = cum; break; }
      cum += h;
    }
    s_coarse = cb; s_above = above;
  }
  __syncthreads();
  const int cb = s_coarse;
  if (cb >= 0) {
    for (int i = tid; i < 1024; i += nt) hist[i] = 0u;
    __syncthreads();
    // pass 2: refine inside the crossing coarse bucket (bits >> 10)
    for (int k = tid; k < NCAND; k += nt) {
      int n = k / NC, c = k - n * NC;
      float s = po[n * 85 + 4] * po[n * 85 + 5 + c];
      unsigned bits = __float_as_uint(s);
      if (s > CONF && (int)(bits >> 20) == cb)
        atomicAdd(&hist[(bits >> 10) & 0x3FFu], 1u);
    }
    __syncthreads();
    if (tid == 0) {
      unsigned cum = s_above; int fb;
      for (fb = 1023; fb >= 0; --fb) { cum += hist[fb]; if (cum >= (unsigned)KPRE) break; }
      if (fb < 0) fb = 0;
      s_thresh = ((unsigned)cb << 20) | ((unsigned)fb << 10);
    }
  } else if (tid == 0) {
    s_thresh = 0x3DCCCCCDu;  // bits(0.1f); float compare gates collection anyway
  }
  __syncthreads();
  if (tid == 0) { thresh[b] = s_thresh; ccount[b] = 0u; }
}

// ---------------------------------------------------------------------------
// Kernel 2: collect candidates >= cutoff (key = score_bits : ~idx)
// ---------------------------------------------------------------------------
__global__ void collect(const float* __restrict__ pred,
                        const unsigned* __restrict__ thresh,
                        unsigned* __restrict__ ccount,
                        unsigned long long* __restrict__ cand) {
  const int b = blockIdx.y;
  const unsigned th = thresh[b];
  const float* po = pred + (size_t)b * NPRED * 85;
  const int stride = gridDim.x * blockDim.x;
  for (int k = blockIdx.x * blockDim.x + threadIdx.x; k < NCAND; k += stride) {
    int n = k / NC, c = k - n * NC;
    float s = po[n * 85 + 4] * po[n * 85 + 5 + c];
    unsigned bits = __float_as_uint(s);
    if (s > CONF && bits >= th) {
      unsigned pos = atomicAdd(&ccount[b], 1u);
      if (pos < CAP)
        cand[(size_t)b * CAP + pos] =
            ((unsigned long long)bits << 32) | (unsigned long long)(0xFFFFFFFFu - (unsigned)k);
    }
  }
}

// ---------------------------------------------------------------------------
// Kernel 3: bitonic sort (desc) + class-aware sequential NMS + compaction
// ---------------------------------------------------------------------------
__global__ void sort_nms(const float* __restrict__ pred,
                         const unsigned long long* __restrict__ cand,
                         const unsigned* __restrict__ ccount,
                         float* __restrict__ boxes, int* __restrict__ nkept) {
  const int b = blockIdx.x, tid = threadIdx.x, nt = blockDim.x;
  __shared__ unsigned long long keys[CAP];
  __shared__ float obx[KPRE][4];   // class-offset xyxy (for IoU)
  __shared__ float rbx[KPRE][4];   // raw xyxy
  __shared__ float area[KPRE];
  __shared__ int keep[KPRE];
  __shared__ int valid[KPRE];
  __shared__ int s_sup;

  unsigned cc = ccount[b];
  int cnt = (cc < (unsigned)CAP) ? (int)cc : CAP;
  for (int i = tid; i < CAP; i += nt)
    keys[i] = (i < cnt) ? cand[(size_t)b * CAP + i] : 0ull;
  __syncthreads();

  // bitonic sort, descending by key
  for (unsigned k = 2; k <= CAP; k <<= 1) {
    for (unsigned j = k >> 1; j > 0; j >>= 1) {
      for (unsigned i = tid; i < CAP; i += nt) {
        unsigned ixj = i ^ j;
        if (ixj > i) {
          bool up = ((i & k) == 0u);
          unsigned long long x = keys[i], y = keys[ixj];
          if (up ? (x < y) : (x > y)) { keys[i] = y; keys[ixj] = x; }
        }
      }
      __syncthreads();
    }
  }

  const int top = (cnt < KPRE) ? cnt : KPRE;
  const float* po = pred + (size_t)b * NPRED * 85;
  for (int i = tid; i < KPRE; i += nt) {
    int ok = (i < top);
    valid[i] = ok; keep[i] = 0;
    float x1 = 0.f, y1 = 0.f, x2 = 0.f, y2 = 0.f, co = 0.f;
    if (ok) {
      unsigned kidx = 0xFFFFFFFFu - (unsigned)(keys[i] & 0xFFFFFFFFull);
      int n = (int)(kidx / NC), c = (int)(kidx - (unsigned)n * NC);
      float cx = po[n * 85 + 0], cy = po[n * 85 + 1];
      float w = po[n * 85 + 2], h = po[n * 85 + 3];
      x1 = cx - 0.5f * w; y1 = cy - 0.5f * h; x2 = cx + 0.5f * w; y2 = cy + 0.5f * h;
      co = (float)c * 4096.0f;
    }
    rbx[i][0] = x1; rbx[i][1] = y1; rbx[i][2] = x2; rbx[i][3] = y2;
    obx[i][0] = x1 + co; obx[i][1] = y1 + co; obx[i][2] = x2 + co; obx[i][3] = y2 + co;
    area[i] = (x2 - x1) * (y2 - y1);
  }
  __syncthreads();

  // serial NMS: step i parallel over kept j < i
  for (int i = 0; i < KPRE; ++i) {
    if (tid == 0) s_sup = 0;
    __syncthreads();
    float ax1 = obx[i][0], ay1 = obx[i][1], ax2 = obx[i][2], ay2 = obx[i][3], aa = area[i];
    int sup = 0;
    for (int j = tid; j < i; j += nt) {
      if (keep[j]) {
        float tlx = fmaxf(ax1, obx[j][0]), tly = fmaxf(ay1, obx[j][1]);
        float brx = fminf(ax2, obx[j][2]), bry = fminf(ay2, obx[j][3]);
        float iw = fmaxf(brx - tlx, 0.f), ih = fmaxf(bry - tly, 0.f);
        float inter = iw * ih;
        if (inter / (aa + area[j] - inter + 1e-7f) > IOUTH) sup = 1;
      }
    }
    if (sup) s_sup = 1;
    __syncthreads();
    if (tid == 0) keep[i] = valid[i] && !s_sup;
    __syncthreads();
  }

  if (tid == 0) {
    int c2 = 0;
    for (int i = 0; i < KPRE && c2 < MAXDET; ++i) {
      if (keep[i]) {
        float* dst = boxes + (size_t)(b * MAXDET + c2) * 4;
        dst[0] = rbx[i][0]; dst[1] = rbx[i][1]; dst[2] = rbx[i][2]; dst[3] = rbx[i][3];
        ++c2;
      }
    }
    nkept[b] = c2;
    for (int i = c2; i < MAXDET; ++i) {
      float* dst = boxes + (size_t)(b * MAXDET + i) * 4;
      dst[0] = dst[1] = dst[2] = dst[3] = 0.f;
    }
  }
}

// ---------------------------------------------------------------------------
// Kernel 4: 1x1 RoIAlign gather over 4 pyramid levels -> f16 feature matrix
// ---------------------------------------------------------------------------
struct Bil { int x0, y0, x1, y1; float w00, w01, w10, w11; };
__device__ inline Bil mkbil(float cx, float cy, float scale, int W, int H) {
  Bil r;
  float x = fminf(fmaxf(cx * scale, 0.f), (float)(W - 1));
  float y = fminf(fmaxf(cy * scale, 0.f), (float)(H - 1));
  r.x0 = (int)floorf(x); r.y0 = (int)floorf(y);
  r.x1 = min(r.x0 + 1, W - 1); r.y1 = min(r.y0 + 1, H - 1);
  float wx = x - (float)r.x0, wy = y - (float)r.y0;
  r.w00 = (1.f - wy) * (1.f - wx); r.w01 = (1.f - wy) * wx;
  r.w10 = wy * (1.f - wx);         r.w11 = wy * wx;
  return r;
}
__device__ inline float bsample(const float* base, int W, const Bil& L) {
  return base[L.y0 * W + L.x0] * L.w00 + base[L.y0 * W + L.x1] * L.w01 +
         base[L.y1 * W + L.x0] * L.w10 + base[L.y1 * W + L.x1] * L.w11;
}

__global__ void roi_gather(const float* __restrict__ f1, const float* __restrict__ f2,
                           const float* __restrict__ f3, const float* __restrict__ f4,
                           const float* __restrict__ boxes, _Float16* __restrict__ fmat) {
  const int row = blockIdx.x;          // 0..799
  const int b = row / MAXDET;
  const float* bx = boxes + (size_t)row * 4;
  float cx = (bx[0] + bx[2]) * 0.5f, cy = (bx[1] + bx[3]) * 0.5f;
  Bil l0 = mkbil(cx, cy, 1.f / 8.f, 64, 64);
  Bil l1 = mkbil(cx, cy, 1.f / 16.f, 32, 32);
  Bil l2 = mkbil(cx, cy, 1.f / 32.f, 16, 16);
  Bil l3 = mkbil(cx, cy, 1.f / 64.f, 8, 8);
  const float* b0 = f1 + (size_t)b * 128 * 64 * 64;
  const float* b1 = f2 + (size_t)b * 256 * 32 * 32;
  const float* b2 = f3 + (size_t)b * 512 * 16 * 16;
  const float* b3 = f4 + (size_t)b * 1024 * 8 * 8;
  _Float16* frow = fmat + (size_t)row * 1920;
  for (int ch = threadIdx.x; ch < 1920; ch += blockDim.x) {
    float v;
    if (ch < 128)       v = bsample(b0 + (size_t)ch * 4096, 64, l0);
    else if (ch < 384)  v = bsample(b1 + (size_t)(ch - 128) * 1024, 32, l1);
    else if (ch < 896)  v = bsample(b2 + (size_t)(ch - 384) * 256, 16, l2);
    else                v = bsample(b3 + (size_t)(ch - 896) * 64, 8, l3);
    frow[ch] = (_Float16)v;
  }
}

// ---------------------------------------------------------------------------
// Kernel 5: convert W1/W2 to f16 in (k,k+1)-pair-major layout for B fragments
// ---------------------------------------------------------------------------
__global__ void cvt_weights(const float* __restrict__ W1, const float* __restrict__ W2,
                            _Float16* __restrict__ W1p, _Float16* __restrict__ W2p) {
  int i = blockIdx.x * blockDim.x + threadIdx.x;
  if (i < 1920 * 64) {
    int k = i >> 6, n = i & 63;
    W1p[((size_t)(k >> 1) * 64 + n) * 2 + (k & 1)] = (_Float16)W1[i];
  }
  if (i < 64 * 64) {
    int k = i >> 6, n = i & 63;
    W2p[((size_t)(k >> 1) * 64 + n) * 2 + (k & 1)] = (_Float16)W2[i];
  }
}

// ---------------------------------------------------------------------------
// Kernel 6: GEMM1 [800,1920]x[1920,64] via v_wmma_f32_16x16x32_f16, LeakyReLU
// one wave32 per 16x16 D tile; 50 M-tiles x 4 N-tiles; K stepped by 32
// ---------------------------------------------------------------------------
__global__ void gemm1(const _Float16* __restrict__ A, const _Float16* __restrict__ Bp,
                      const float* __restrict__ bias, _Float16* __restrict__ H1) {
  const int tile = blockIdx.x, tm = tile >> 2, tn = tile & 3;
  const int lane = threadIdx.x, half = lane >> 4, l15 = lane & 15;
  const int mrow = tm * 16 + l15, ncol = tn * 16 + l15;
  v8f acc = {};
  for (int k0 = 0; k0 < 1920; k0 += 32) {
    union { v16h v; unsigned u[8]; } a, bf;
#pragma unroll
    for (int q = 0; q < 8; ++q) {
      int ka = k0 + ((q & 3) * 2) + half * 8 + ((q >> 2) * 16);   // A 16x32 f16 layout
      a.u[q] = *(const unsigned*)(A + (size_t)mrow * 1920 + ka);
      int kb = k0 + ((q & 3) * 2) + half * 8 + ((q >> 2) * 16);   // B 32x16 f16 layout
      bf.u[q] = *(const unsigned*)(Bp + ((size_t)(kb >> 1) * 64 + ncol) * 2);
    }
    acc = __builtin_amdgcn_wmma_f32_16x16x32_f16(false, a.v, false, bf.v,
                                                 (short)0, acc, false, false);
  }
#pragma unroll
  for (int r = 0; r < 8; ++r) {
    int gm = tm * 16 + r + 8 * half;
    float val = acc[r] + bias[ncol];
    val = val > 0.f ? val : 0.01f * val;
    H1[(size_t)gm * 64 + ncol] = (_Float16)val;
  }
}

// ---------------------------------------------------------------------------
// Kernel 7: GEMM2 [800,64]x[64,64] via WMMA; LeakyReLU + mask; writes d_out
// ---------------------------------------------------------------------------
__global__ void gemm2(const _Float16* __restrict__ H1, const _Float16* __restrict__ W2p,
                      const float* __restrict__ bias, const int* __restrict__ nkept,
                      float* __restrict__ outp) {
  const int tile = blockIdx.x, tm = tile >> 2, tn = tile & 3;
  const int lane = threadIdx.x, half = lane >> 4, l15 = lane & 15;
  const int mrow = tm * 16 + l15, ncol = tn * 16 + l15;
  v8f acc = {};
  for (int k0 = 0; k0 < 64; k0 += 32) {
    union { v16h v; unsigned u[8]; } a, bf;
#pragma unroll
    for (int q = 0; q < 8; ++q) {
      int ka = k0 + ((q & 3) * 2) + half * 8 + ((q >> 2) * 16);
      a.u[q] = *(const unsigned*)(H1 + (size_t)mrow * 64 + ka);
      int kb = k0 + ((q & 3) * 2) + half * 8 + ((q >> 2) * 16);
      bf.u[q] = *(const unsigned*)(W2p + ((size_t)(kb >> 1) * 64 + ncol) * 2);
    }
    acc = __builtin_amdgcn_wmma_f32_16x16x32_f16(false, a.v, false, bf.v,
                                                 (short)0, acc, false, false);
  }
#pragma unroll
  for (int r = 0; r < 8; ++r) {
    int gm = tm * 16 + r + 8 * half;
    float val = acc[r] + bias[ncol];
    val = val > 0.f ? val : 0.01f * val;
    int bb = gm / MAXDET, mm = gm - bb * MAXDET;
    float mk = (mm < nkept[bb]) ? 1.f : 0.f;
    outp[(size_t)gm * 68 + 4 + ncol] = mk * val;
  }
}

// ---------------------------------------------------------------------------
// Kernel 8: normalized box part of the output (cols 0..3)
// ---------------------------------------------------------------------------
__global__ void boxes_out(const float* __restrict__ boxes, const int* __restrict__ nkept,
                          float* __restrict__ outp) {
  int i = blockIdx.x * blockDim.x + threadIdx.x;
  if (i >= BATCH * MAXDET * 4) return;
  int row = i >> 2, j = i & 3;
  int b = row / MAXDET, m = row - b * MAXDET;
  float mk = (m < nkept[b]) ? 1.f : 0.f;
  outp[(size_t)row * 68 + j] = mk * boxes[i] * (1.0f / 512.0f);
}

// ---------------------------------------------------------------------------
extern "C" void kernel_launch(void* const* d_in, const int* in_sizes, int n_in,
                              void* d_out, int out_size, void* d_ws, size_t ws_size,
                              hipStream_t stream) {
  (void)in_sizes; (void)n_in; (void)out_size; (void)ws_size;
  const float* pred = (const float*)d_in[0];
  const float* f1   = (const float*)d_in[1];
  const float* f2   = (const float*)d_in[2];
  const float* f3   = (const float*)d_in[3];
  const float* f4   = (const float*)d_in[4];
  const float* W1   = (const float*)d_in[5];
  const float* b1   = (const float*)d_in[6];
  const float* W2   = (const float*)d_in[7];
  const float* b2   = (const float*)d_in[8];
  float* outp = (float*)d_out;

  char* w = (char*)d_ws;
  size_t off = 0;
  auto take = [&](size_t bytes) { char* p = w + off; off = (off + bytes + 255) & ~(size_t)255; return p; };
  unsigned* thresh            = (unsigned*)take(BATCH * 4);
  unsigned* ccount            = (unsigned*)take(BATCH * 4);
  int* nkept                  = (int*)take(BATCH * 4);
  unsigned long long* cand    = (unsigned long long*)take((size_t)BATCH * CAP * 8);
  float* boxes                = (float*)take((size_t)BATCH * MAXDET * 4 * 4);
  _Float16* fmat              = (_Float16*)take((size_t)BATCH * MAXDET * 1920 * 2);
  _Float16* W1p               = (_Float16*)take((size_t)1920 * 64 * 2);
  _Float16* W2p               = (_Float16*)take((size_t)64 * 64 * 2);
  _Float16* h1                = (_Float16*)take((size_t)BATCH * MAXDET * 64 * 2);

  topk_thresh<<<BATCH, 1024, 0, stream>>>(pred, thresh, ccount);
  collect<<<dim3(128, BATCH), 256, 0, stream>>>(pred, thresh, ccount, cand);
  sort_nms<<<BATCH, 512, 0, stream>>>(pred, cand, ccount, boxes, nkept);
  cvt_weights<<<(1920 * 64 + 255) / 256, 256, 0, stream>>>(W1, W2, W1p, W2p);
  roi_gather<<<BATCH * MAXDET, 256, 0, stream>>>(f1, f2, f3, f4, boxes, fmat);
  gemm1<<<(BATCH * MAXDET / 16) * 4, 32, 0, stream>>>(fmat, W1p, b1, h1);
  gemm2<<<(BATCH * MAXDET / 16) * 4, 32, 0, stream>>>(h1, W2p, b2, nkept, outp);
  boxes_out<<<(BATCH * MAXDET * 4 + 255) / 256, 256, 0, stream>>>(boxes, nkept, outp);
}
